// IKG_Enhancer_22874995818772
// MI455X (gfx1250) — matrix operand order repaired and checked by reading, not compiled, not run on hardware
//
#include <hip/hip_runtime.h>
#include <hip/hip_bf16.h>

typedef __attribute__((ext_vector_type(16))) _Float16 v16h;
typedef __attribute__((ext_vector_type(8)))  float    v8f;

#define NEG_SLOPE 0.2f
#define ATTN_CH 4096

// ---------------------------------------------------------------------------
// float atomic max via int/uint punning (works for mixed signs, init -FLT_MAX)
// ---------------------------------------------------------------------------
__device__ __forceinline__ void atomicMaxFloat(float* addr, float val) {
    if (val >= 0.0f) {
        atomicMax((int*)addr, __float_as_int(val));
    } else {
        atomicMin((unsigned int*)addr, __float_as_uint(val));
    }
}

// ---------------------------------------------------------------------------
// Weight conversion: f32 -> f16, packed into WMMA B-fragment-ready layout.
// Fragment layout (per matrix, K rows x Ncols cols, NT = Ncols/16):
//   Bf[(((kt*NT + nt)*32 + lane)*16 + j] = B[kt*32 + 16*(lane>>4) + j][nt*16 + (lane&15)]
// so each lane's 16 halves for one (ktile, ntile) are contiguous (32 B).
// ---------------------------------------------------------------------------
__global__ void conv_weights_kernel(const float* __restrict__ gcnK_W,
                                    const float* __restrict__ gcnV_W,
                                    const float* __restrict__ mha_in_w,
                                    const float* __restrict__ gat_W,
                                    const float* __restrict__ t1_W,
                                    const float* __restrict__ t2_W,
                                    _Float16* __restrict__ wK, _Float16* __restrict__ wV,
                                    _Float16* __restrict__ wkT, _Float16* __restrict__ wvT,
                                    _Float16* __restrict__ wg, _Float16* __restrict__ w1,
                                    _Float16* __restrict__ w2) {
    int i = blockIdx.x * blockDim.x + threadIdx.x;
    const int j    = i & 15;
    const int lane = (i >> 4) & 31;
    const int tile = i >> 9;
    const int half = lane >> 4;
    const int r    = lane & 15;
    if (i < 4096) {
        // 64x64 matrices, NT = 4
        int kt = tile >> 2, nt = tile & 3;
        int k = kt * 32 + 16 * half + j;
        int n = nt * 16 + r;
        wK[i] = (_Float16)gcnK_W[k * 64 + n];
        wV[i] = (_Float16)gcnV_W[k * 64 + n];
        wg[i] = (_Float16)gat_W[k * 64 + n];
        // kh = x_K @ Wk.T  =>  B[k][n] = in_w[(64+n)*64 + k]
        wkT[i] = (_Float16)mha_in_w[(64 + n) * 64 + k];
        wvT[i] = (_Float16)mha_in_w[(128 + n) * 64 + k];
    }
    if (i < 8192) {
        // t1: K=64, Ncols=128, NT=8
        {
            int kt = tile >> 3, nt = tile & 7;
            int k = kt * 32 + 16 * half + j;
            int n = nt * 16 + r;
            w1[i] = (_Float16)t1_W[k * 128 + n];
        }
        // t2: K=128, Ncols=64, NT=4
        {
            int kt = tile >> 2, nt = tile & 3;
            int k = kt * 32 + 16 * half + j;
            int n = nt * 16 + r;
            w2[i] = (_Float16)t2_W[k * 64 + n];
        }
    }
}

// ---------------------------------------------------------------------------
// WMMA GEMM: C[M,Ncols] = A[M,K](f32, converted on load) * Bf(f16 fragments)
//            + bias, optional relu.
// Template <K, NT>: K-loop fully unrolled (2-4 static v_wmma), Ncols folds
// into immediate store offsets. blockDim = 32*NT, grid.x = ceil(M/16).
// Wave w computes the 16x16 tile at columns [w*16, w*16+16).
// EXEC stays full through the WMMAs (clamped loads, predicated tail stores).
// ---------------------------------------------------------------------------
template <int K, int NT>
__global__ void gemm_wmma_kernel(const float* __restrict__ A,
                                 const _Float16* __restrict__ Bf,
                                 const float* __restrict__ bias,
                                 float* __restrict__ C,
                                 int M, int relu) {
    constexpr int Ncols = NT * 16;
    const int wave = threadIdx.x >> 5;
    const int lane = threadIdx.x & 31;
    const int row0 = blockIdx.x * 16;
    const int r    = lane & 15;
    const int half = lane >> 4;

    int arow = row0 + r;
    if (arow >= M) arow = M - 1;  // clamp: keep EXEC all-ones for WMMA

    const float*    abase = A + (size_t)arow * K + 16 * half;
    const _Float16* bbase = Bf + ((size_t)wave * 32 + lane) * 16;

    v8f acc = {};
#pragma unroll
    for (int kt = 0; kt < K / 32; ++kt) {
        // A fragment (16-bit A 16x32): half h holds K = 16h+0..15
        const float* ap = abase + kt * 32;
        v16h af;
#pragma unroll
        for (int j = 0; j < 8; ++j) af[j]     = (_Float16)ap[j];
#pragma unroll
        for (int j = 0; j < 8; ++j) af[8 + j] = (_Float16)ap[8 + j];
        // B fragment: one contiguous 32-byte load
        v16h bf = *(const v16h*)(bbase + (size_t)kt * NT * 32 * 16);
        acc = __builtin_amdgcn_wmma_f32_16x16x32_f16(
            false, af, false, bf, (short)0, acc, false, false);
    }

    // C layout: VGPR v -> row = row0 + v + 8*half, col = wave*16 + (lane&15)
    const int col = wave * 16 + r;
    const float bv = bias ? bias[col] : 0.0f;
    float* cbase = C + (size_t)(row0 + 8 * half) * Ncols + col;
    if (row0 + 16 <= M) {  // full tile: branch-free, immediate-offset stores
#pragma unroll
        for (int v = 0; v < 8; ++v) {
            float val = acc[v] + bv;
            if (relu) val = fmaxf(val, 0.0f);
            cbase[(size_t)v * Ncols] = val;
        }
    } else {               // tail tile
#pragma unroll
        for (int v = 0; v < 8; ++v) {
            if (row0 + 8 * half + v < M) {
                float val = acc[v] + bv;
                if (relu) val = fmaxf(val, 0.0f);
                cbase[(size_t)v * Ncols] = val;
            }
        }
    }
}

// ---------------------------------------------------------------------------
// Degree / norm
// ---------------------------------------------------------------------------
__global__ void deg_init_kernel(float* __restrict__ deg, int n) {
    int i = blockIdx.x * blockDim.x + threadIdx.x;
    if (i < n) deg[i] = 1.0f;  // self loop
}
__global__ void deg_count_kernel(const int* __restrict__ dst, float* __restrict__ deg, int e) {
    int i = blockIdx.x * blockDim.x + threadIdx.x;
    if (i < e) atomicAdd(&deg[dst[i]], 1.0f);
}
__global__ void dinv_kernel(float* __restrict__ deg, int n) {
    int i = blockIdx.x * blockDim.x + threadIdx.x;
    if (i < n) {
        float d = deg[i];
        deg[i] = (d > 0.0f) ? rsqrtf(d) : 0.0f;  // in-place -> dinv
    }
}

// ---------------------------------------------------------------------------
// GCN aggregation: self-loop + bias init, then wave-per-edge atomic scatter
// ---------------------------------------------------------------------------
__global__ void agg_init_kernel(const float* __restrict__ hK, const float* __restrict__ hV,
                                const float* __restrict__ dinv,
                                const float* __restrict__ bK, const float* __restrict__ bV,
                                float* __restrict__ aggK, float* __restrict__ aggV, int n) {
    size_t i = (size_t)blockIdx.x * blockDim.x + threadIdx.x;
    size_t tot = (size_t)n * 64;
    if (i < tot) {
        int node = (int)(i >> 6), d = (int)(i & 63);
        float w = dinv[node] * dinv[node];
        aggK[i] = hK[i] * w + bK[d];
        aggV[i] = hV[i] * w + bV[d];
    }
}

__global__ void agg_edges_kernel(const int* __restrict__ src, const int* __restrict__ dst,
                                 const float* __restrict__ dinv,
                                 const float* __restrict__ hK, const float* __restrict__ hV,
                                 float* __restrict__ aggK, float* __restrict__ aggV, int e) {
    int wid  = (int)(((size_t)blockIdx.x * blockDim.x + threadIdx.x) >> 5);
    int lane = threadIdx.x & 31;
    if (wid >= e) return;
    int s = src[wid], d = dst[wid];
    float w = dinv[s] * dinv[d];
    size_t sb = (size_t)s * 64, db = (size_t)d * 64;
    atomicAdd(&aggK[db + lane],      w * hK[sb + lane]);
    atomicAdd(&aggK[db + lane + 32], w * hK[sb + lane + 32]);
    atomicAdd(&aggV[db + lane],      w * hV[sb + lane]);
    atomicAdd(&aggV[db + lane + 32], w * hV[sb + lane + 32]);
}

// ---------------------------------------------------------------------------
// MHA: tiny Q projection, then flash-style two-pass softmax over 100k keys
// ---------------------------------------------------------------------------
__global__ void qh_kernel(const float* __restrict__ superQ, const float* __restrict__ in_w,
                          const float* __restrict__ in_b, float* __restrict__ qh) {
    int i = blockIdx.x * blockDim.x + threadIdx.x;
    if (i >= 640) return;
    int l = i >> 6, d = i & 63;
    float acc = in_b[d];
    for (int k = 0; k < 64; ++k) acc += superQ[l * 64 + k] * in_w[d * 64 + k];
    qh[i] = acc * 0.35355339059327373f;  // 1/sqrt(8)
}

__global__ void attn_max_kernel(const float* __restrict__ qh, const float* __restrict__ kh,
                                float* __restrict__ pmax, int n, int nch) {
    int hl = blockIdx.y;  // hl = l*8 + h
    int l = hl >> 3, h = hl & 7;
    const float* q = qh + l * 64 + h * 8;
    float q0=q[0],q1=q[1],q2=q[2],q3=q[3],q4=q[4],q5=q[5],q6=q[6],q7=q[7];
    float lm = -3.402823466e38f;
    int start = blockIdx.x * ATTN_CH;
    int end = start + ATTN_CH; if (end > n) end = n;
    for (int i = start + threadIdx.x; i < end; i += blockDim.x) {
        const float* kp = kh + (size_t)i * 64 + h * 8;
        float lg = q0*kp[0]+q1*kp[1]+q2*kp[2]+q3*kp[3]+q4*kp[4]+q5*kp[5]+q6*kp[6]+q7*kp[7];
        lm = fmaxf(lm, lg);
    }
    __shared__ float red[256];
    red[threadIdx.x] = lm;
    __syncthreads();
    for (int s = 128; s > 0; s >>= 1) {
        if (threadIdx.x < s) red[threadIdx.x] = fmaxf(red[threadIdx.x], red[threadIdx.x + s]);
        __syncthreads();
    }
    if (threadIdx.x == 0) pmax[hl * nch + blockIdx.x] = red[0];
}

__global__ void attn_reduce_kernel(const float* __restrict__ pmax, float* __restrict__ m,
                                   float* __restrict__ s, float* __restrict__ o, int nch) {
    int hl = threadIdx.x;
    if (hl < 80) {
        float mm = -3.402823466e38f;
        for (int j = 0; j < nch; ++j) mm = fmaxf(mm, pmax[hl * nch + j]);
        m[hl] = mm;
        s[hl] = 0.0f;
        for (int j = 0; j < 8; ++j) o[hl * 8 + j] = 0.0f;
    }
}

__global__ void attn_acc_kernel(const float* __restrict__ qh, const float* __restrict__ kh,
                                const float* __restrict__ vh, const float* __restrict__ m,
                                float* __restrict__ s, float* __restrict__ o, int n) {
    int hl = blockIdx.y;
    int l = hl >> 3, h = hl & 7;
    const float* q = qh + l * 64 + h * 8;
    float q0=q[0],q1=q[1],q2=q[2],q3=q[3],q4=q[4],q5=q[5],q6=q[6],q7=q[7];
    float mm = m[hl];
    float ls = 0.0f;
    float lo[8] = {0.f,0.f,0.f,0.f,0.f,0.f,0.f,0.f};
    int start = blockIdx.x * ATTN_CH;
    int end = start + ATTN_CH; if (end > n) end = n;
    for (int i = start + threadIdx.x; i < end; i += blockDim.x) {
        const float* kp = kh + (size_t)i * 64 + h * 8;
        const float* vp = vh + (size_t)i * 64 + h * 8;
        // deep prefetch of next iteration's K row (global_prefetch_b8)
        __builtin_prefetch((const void*)(kp + (size_t)64 * 256), 0, 0);
        float lg = q0*kp[0]+q1*kp[1]+q2*kp[2]+q3*kp[3]+q4*kp[4]+q5*kp[5]+q6*kp[6]+q7*kp[7];
        float ee = __expf(lg - mm);
        ls += ee;
#pragma unroll
        for (int j = 0; j < 8; ++j) lo[j] += ee * vp[j];
    }
    // wave32 reduction
    for (int sft = 16; sft > 0; sft >>= 1) {
        ls += __shfl_xor(ls, sft);
#pragma unroll
        for (int j = 0; j < 8; ++j) lo[j] += __shfl_xor(lo[j], sft);
    }
    if ((threadIdx.x & 31) == 0) {
        atomicAdd(&s[hl], ls);
#pragma unroll
        for (int j = 0; j < 8; ++j) atomicAdd(&o[hl * 8 + j], lo[j]);
    }
}

// out-proj + mean-context + a_mod = a_src + context @ Wc   (one 64-thread block)
__global__ void attn_final_kernel(const float* __restrict__ s, const float* __restrict__ o,
                                  const float* __restrict__ out_w, const float* __restrict__ out_b,
                                  const float* __restrict__ gat_Wc, const float* __restrict__ a_src,
                                  float* __restrict__ a_mod) {
    __shared__ float ao[640];
    __shared__ float ctx[64];
    int t = threadIdx.x;  // 64 threads
    for (int i = t; i < 640; i += 64) {
        int l = i / 64, d = i % 64;
        int h = d >> 3, j = d & 7;
        int hl = l * 8 + h;
        float den = s[hl];
        ao[i] = o[hl * 8 + j] / (den + 1e-16f);
    }
    __syncthreads();
    float cx = 0.0f;
    for (int l = 0; l < 10; ++l) {
        float acc = out_b[t];
        for (int k = 0; k < 64; ++k) acc += ao[l * 64 + k] * out_w[t * 64 + k];
        cx += acc;
    }
    ctx[t] = cx * 0.1f;  // mean over 10 clusters
    __syncthreads();
    float am = a_src[t];
    for (int k = 0; k < 64; ++k) am += ctx[k] * gat_Wc[k * 64 + t];
    a_mod[t] = am;
}

// ---------------------------------------------------------------------------
// Context GAT
// ---------------------------------------------------------------------------
__global__ void gat_scores_kernel(const float* __restrict__ hg, const float* __restrict__ a_mod,
                                  const float* __restrict__ a_dst,
                                  float* __restrict__ ssrc, float* __restrict__ sdst, int n) {
    int wid  = (int)(((size_t)blockIdx.x * blockDim.x + threadIdx.x) >> 5);
    int lane = threadIdx.x & 31;
    if (wid >= n) return;
    const float* hp = hg + (size_t)wid * 64;
    float v1 = hp[lane] * a_mod[lane] + hp[lane + 32] * a_mod[lane + 32];
    float v2 = hp[lane] * a_dst[lane] + hp[lane + 32] * a_dst[lane + 32];
    for (int sft = 16; sft > 0; sft >>= 1) {
        v1 += __shfl_xor(v1, sft);
        v2 += __shfl_xor(v2, sft);
    }
    if (lane == 0) { ssrc[wid] = v1; sdst[wid] = v2; }
}

__global__ void gat_init_kernel(float* __restrict__ num, float* __restrict__ gmax,
                                float* __restrict__ den, int n) {
    size_t i = (size_t)blockIdx.x * blockDim.x + threadIdx.x;
    size_t tot = (size_t)n * 64;
    if (i < tot) num[i] = 0.0f;
    if (i < (size_t)n) { gmax[i] = -3.402823466e38f; den[i] = 0.0f; }
}

__global__ void gat_logit_max_kernel(const int* __restrict__ src, const int* __restrict__ dst,
                                     const float* __restrict__ ssrc, const float* __restrict__ sdst,
                                     float* __restrict__ gmax, int e) {
    int i = blockIdx.x * blockDim.x + threadIdx.x;
    if (i >= e) return;
    float lg = ssrc[src[i]] + sdst[dst[i]];
    lg = (lg > 0.0f) ? lg : NEG_SLOPE * lg;
    atomicMaxFloat(&gmax[dst[i]], lg);
}

__global__ void gat_acc_kernel(const int* __restrict__ src, const int* __restrict__ dst,
                               const float* __restrict__ ssrc, const float* __restrict__ sdst,
                               const float* __restrict__ gmax, const float* __restrict__ hg,
                               float* __restrict__ num, float* __restrict__ den, int e) {
    int wid  = (int)(((size_t)blockIdx.x * blockDim.x + threadIdx.x) >> 5);
    int lane = threadIdx.x & 31;
    if (wid >= e) return;
    int si = src[wid], di = dst[wid];
    float lg = ssrc[si] + sdst[di];
    lg = (lg > 0.0f) ? lg : NEG_SLOPE * lg;
    float ee = __expf(lg - gmax[di]);
    size_t sb = (size_t)si * 64, db = (size_t)di * 64;
    atomicAdd(&num[db + lane],      ee * hg[sb + lane]);
    atomicAdd(&num[db + lane + 32], ee * hg[sb + lane + 32]);
    if (lane == 0) atomicAdd(&den[di], ee);
}

__global__ void gat_final_kernel(const float* __restrict__ num, const float* __restrict__ den,
                                 const float* __restrict__ b, float* __restrict__ g, int n) {
    size_t i = (size_t)blockIdx.x * blockDim.x + threadIdx.x;
    size_t tot = (size_t)n * 64;
    if (i < tot) {
        int node = (int)(i >> 6), d = (int)(i & 63);
        g[i] = num[i] / (den[node] + 1e-16f) + b[d];
    }
}

// ---------------------------------------------------------------------------
// Launch
// ---------------------------------------------------------------------------
extern "C" void kernel_launch(void* const* d_in, const int* in_sizes, int n_in,
                              void* d_out, int out_size, void* d_ws, size_t ws_size,
                              hipStream_t stream) {
    const float* x        = (const float*)d_in[0];
    const int*   ei       = (const int*)d_in[1];
    const float* gcnK_W   = (const float*)d_in[2];
    const float* gcnK_b   = (const float*)d_in[3];
    const float* gcnV_W   = (const float*)d_in[4];
    const float* gcnV_b   = (const float*)d_in[5];
    const float* super_Q  = (const float*)d_in[6];
    const float* mha_in_w = (const float*)d_in[7];
    const float* mha_in_b = (const float*)d_in[8];
    const float* mha_out_w= (const float*)d_in[9];
    const float* mha_out_b= (const float*)d_in[10];
    const float* gat_W    = (const float*)d_in[11];
    const float* gat_Wc   = (const float*)d_in[12];
    const float* gat_asrc = (const float*)d_in[13];
    const float* gat_adst = (const float*)d_in[14];
    const float* gat_b    = (const float*)d_in[15];
    const float* t1_W     = (const float*)d_in[16];
    const float* t1_b     = (const float*)d_in[17];
    const float* t2_W     = (const float*)d_in[18];
    const float* t2_b     = (const float*)d_in[19];

    const int Nn = in_sizes[0] / 64;
    const int Ee = in_sizes[1] / 2;
    const int* srcp = ei;
    const int* dstp = ei + Ee;

    // -------- workspace layout --------
    const size_t NB = (size_t)Nn * 64 * sizeof(float);
    char* w = (char*)d_ws;
    float* bufA  = (float*)(w + 0 * NB);   // hK -> kh -> g
    float* bufB  = (float*)(w + 1 * NB);   // hV -> vh
    float* bufC  = (float*)(w + 2 * NB);   // x_K -> hg
    float* bufD  = (float*)(w + 3 * NB);   // x_V -> num
    float* bufH1 = (float*)(w + 4 * NB);   // h1 [N,128] (2*NB)
    char* sm = w + 6 * NB;
    float* dinv = (float*)sm;              sm += (size_t)Nn * 4;
    float* ssrc = (float*)sm;              sm += (size_t)Nn * 4;
    float* sdst = (float*)sm;              sm += (size_t)Nn * 4;
    float* gmax = (float*)sm;              sm += (size_t)Nn * 4;
    float* den  = (float*)sm;              sm += (size_t)Nn * 4;
    float* qh   = (float*)sm;              sm += 640 * 4;
    float* pmax = (float*)sm;              sm += 80 * 64 * 4;
    float* m80  = (float*)sm;              sm += 80 * 4;
    float* s80  = (float*)sm;              sm += 80 * 4;
    float* o640 = (float*)sm;              sm += 640 * 4;
    float* amod = (float*)sm;              sm += 64 * 4;
    _Float16* wK  = (_Float16*)sm;         sm += 4096 * 2;
    _Float16* wV  = (_Float16*)sm;         sm += 4096 * 2;
    _Float16* wkT = (_Float16*)sm;         sm += 4096 * 2;
    _Float16* wvT = (_Float16*)sm;         sm += 4096 * 2;
    _Float16* wg  = (_Float16*)sm;         sm += 4096 * 2;
    _Float16* w1  = (_Float16*)sm;         sm += 8192 * 2;
    _Float16* w2  = (_Float16*)sm;         sm += 8192 * 2;

    const int TB = 256;
    const int mtiles     = (Nn + 15) / 16;
    const int nodeElems  = Nn * 64;
    const int edgeWaveBl = (Ee * 32 + TB - 1) / TB;
    const int nodeWaveBl = (Nn * 32 + TB - 1) / TB;

    // 0) weights -> f16 fragment-packed
    conv_weights_kernel<<<(8192 + TB - 1) / TB, TB, 0, stream>>>(
        gcnK_W, gcnV_W, mha_in_w, gat_W, t1_W, t2_W, wK, wV, wkT, wvT, wg, w1, w2);

    // 1) degree / dinv
    deg_init_kernel<<<(Nn + TB - 1) / TB, TB, 0, stream>>>(dinv, Nn);
    deg_count_kernel<<<(Ee + TB - 1) / TB, TB, 0, stream>>>(dstp, dinv, Ee);
    dinv_kernel<<<(Nn + TB - 1) / TB, TB, 0, stream>>>(dinv, Nn);

    // 2) GCN projections (WMMA): hK = x@Wk, hV = x@Wv
    gemm_wmma_kernel<64, 4><<<mtiles, 128, 0, stream>>>(x, wK, nullptr, bufA, Nn, 0);
    gemm_wmma_kernel<64, 4><<<mtiles, 128, 0, stream>>>(x, wV, nullptr, bufB, Nn, 0);

    // 3) symmetric-normalized aggregation (self-loop + bias init, edge scatter)
    agg_init_kernel<<<(nodeElems + TB - 1) / TB, TB, 0, stream>>>(
        bufA, bufB, dinv, gcnK_b, gcnV_b, bufC, bufD, Nn);
    agg_edges_kernel<<<edgeWaveBl, TB, 0, stream>>>(
        srcp, dstp, dinv, bufA, bufB, bufC, bufD, Ee);

    // 4) MHA projections (WMMA): kh -> bufA, vh -> bufB
    qh_kernel<<<3, TB, 0, stream>>>(super_Q, mha_in_w, mha_in_b, qh);
    gemm_wmma_kernel<64, 4><<<mtiles, 128, 0, stream>>>(bufC, wkT, mha_in_b + 64,  bufA, Nn, 0);
    gemm_wmma_kernel<64, 4><<<mtiles, 128, 0, stream>>>(bufD, wvT, mha_in_b + 128, bufB, Nn, 0);

    // 5) attention over 100k keys: two-pass online softmax
    const int nch = (Nn + ATTN_CH - 1) / ATTN_CH;
    dim3 agrid(nch, 80);
    attn_max_kernel<<<agrid, 256, 0, stream>>>(qh, bufA, pmax, Nn, nch);
    attn_reduce_kernel<<<1, 128, 0, stream>>>(pmax, m80, s80, o640, nch);
    attn_acc_kernel<<<agrid, 256, 0, stream>>>(qh, bufA, bufB, m80, s80, o640, Nn);
    attn_final_kernel<<<1, 64, 0, stream>>>(s80, o640, mha_out_w, mha_out_b,
                                            gat_Wc, gat_asrc, amod);

    // 6) GAT: hg = x_V @ gat_W (WMMA) -> bufC
    gemm_wmma_kernel<64, 4><<<mtiles, 128, 0, stream>>>(bufD, wg, nullptr, bufC, Nn, 0);
    gat_scores_kernel<<<nodeWaveBl, TB, 0, stream>>>(bufC, amod, gat_adst, ssrc, sdst, Nn);
    gat_init_kernel<<<(nodeElems + TB - 1) / TB, TB, 0, stream>>>(bufD, gmax, den, Nn);
    gat_logit_max_kernel<<<(Ee + TB - 1) / TB, TB, 0, stream>>>(srcp, dstp, ssrc, sdst, gmax, Ee);
    gat_acc_kernel<<<edgeWaveBl, TB, 0, stream>>>(srcp, dstp, ssrc, sdst, gmax, bufC, bufD, den, Ee);
    gat_final_kernel<<<(nodeElems + TB - 1) / TB, TB, 0, stream>>>(bufD, den, gat_b, bufA, Nn);

    // 7) MLP (WMMA): h1 = relu(g@t1+b1) [N,128]; out = relu(h1@t2+b2) [N,64]
    gemm_wmma_kernel<64, 8><<<mtiles, 256, 0, stream>>>(bufA, w1, t1_b, bufH1, Nn, 1);
    gemm_wmma_kernel<128, 4><<<mtiles, 128, 0, stream>>>(bufH1, w2, t2_b, (float*)d_out, Nn, 1);
}
